// Sparse_Attn_Logic_82205674045468
// MI455X (gfx1250) — matrix operand order repaired and checked by reading, not compiled, not run on hardware
//
#include <hip/hip_runtime.h>
#include <hip/hip_fp16.h>

typedef __attribute__((ext_vector_type(16))) _Float16 v16h;
typedef __attribute__((ext_vector_type(8)))  _Float16 v8h;
typedef __attribute__((ext_vector_type(8)))  float    v8f;

#define BATCH 1024
#define CCH   32
#define LIN   1024
#define LCONV 1041
#define LPOOL 520
#define LS1   257
#define LS2   125
#define LS3   31
#define LS4   12

__device__ __forceinline__ float relu_(float x) { return x > 0.f ? x : 0.f; }

// ---- Laplace filter bank: filt[c][k], c in [0,32), k in [0,16) -------------
__device__ __forceinline__ void compute_filt(float* filt, const float* la_a,
                                             const float* la_b, int t, int nth) {
  const float W50  = 314.1592653589793f;            // 2*pi*50
  const float EPQ  = 0.03f / sqrtf(1.0f - 0.0009f); // ep / sqrt(1-ep^2)
  for (int i = t; i < CCH * 16; i += nth) {
    int c = i >> 4, k = i & 15;
    float ratio = la_b[c] / la_a[c];
    float p   = (float)k * (1.0f / 15.0f) - ratio;
    float arg = W50 * (p - 0.1f);
    filt[i] = 0.08f * expf(-EPQ * arg) * (-sinf(arg));
  }
}

// ---- zero stats accumulators ----------------------------------------------
__global__ void k_zero(float* p, int n) {
  int i = blockIdx.x * blockDim.x + threadIdx.x;
  if (i < n) p[i] = 0.f;
}

// ---- pass 1: conv + bias, accumulate per-channel sum / sumsq ---------------
__global__ void k_conv_stats(const float* __restrict__ x, const float* __restrict__ la_a,
                             const float* __restrict__ la_b, const float* __restrict__ la_bias,
                             float* __restrict__ stats) {
  __shared__ float filt[CCH * 16];
  __shared__ float xs[144];
  __shared__ float wb[CCH], csum[CCH], csq[CCH];
  int b = blockIdx.x, n0 = blockIdx.y * 128;
  int t = threadIdx.x;
  compute_filt(filt, la_a, la_b, t, 256);
  if (t < CCH) { wb[t] = la_bias[t]; csum[t] = 0.f; csq[t] = 0.f; }
  for (int j = t; j < 144; j += 256) {
    int ix = n0 - 16 + j;
    xs[j] = (ix >= 0 && ix < LIN) ? x[b * LIN + ix] : 0.f;
  }
  __syncthreads();
  int c = t >> 3;                // 8 threads per channel
  int nlb = (t & 7) * 16;        // 16 outputs each
  float ls = 0.f, lq = 0.f;
  for (int j = 0; j < 16; ++j) {
    int nl = nlb + j, n = n0 + nl;
    if (n < LCONV) {
      float acc = 0.f;
#pragma unroll
      for (int k = 0; k < 16; ++k) acc += filt[c * 16 + k] * xs[nl + k];
      float v = acc + wb[c];
      ls += v; lq += v * v;
    }
  }
  atomicAdd(&csum[c], ls);
  atomicAdd(&csq[c], lq);
  __syncthreads();
  if (t < CCH) { atomicAdd(&stats[t], csum[t]); atomicAdd(&stats[CCH + t], csq[t]); }
}

// ---- finalize BN: scale = g*rsqrt(var+eps), shift = b - mean*scale ---------
__global__ void k_bn_finalize(const float* __restrict__ stats, const float* __restrict__ g,
                              const float* __restrict__ be, float* __restrict__ params,
                              float invN) {
  int c = threadIdx.x;
  if (c < CCH) {
    float mean = stats[c] * invN;
    float var  = stats[CCH + c] * invN - mean * mean;
    float sc   = g[c] * rsqrtf(var + 1e-5f);
    params[c] = sc;
    params[CCH + c] = be[c] - mean * sc;
  }
}

// ---- pass 2: conv + bias + BN1 + avgpool(3,2) -> h (B,32,520) --------------
__global__ void k_conv_bn_pool(const float* __restrict__ x, const float* __restrict__ la_a,
                               const float* __restrict__ la_b, const float* __restrict__ la_bias,
                               const float* __restrict__ bn1, float* __restrict__ h) {
  __shared__ float filt[CCH * 16];
  __shared__ float xs[280];
  __shared__ float sc[CCH], sh[CCH], wb[CCH];
  int b = blockIdx.x, m0 = blockIdx.y * 130;
  int t = threadIdx.x;
  compute_filt(filt, la_a, la_b, t, 256);
  if (t < CCH) { sc[t] = bn1[t]; sh[t] = bn1[CCH + t]; wb[t] = la_bias[t]; }
  for (int j = t; j < 276; j += 256) {
    int ix = 2 * m0 - 16 + j;
    xs[j] = (ix >= 0 && ix < LIN) ? x[b * LIN + ix] : 0.f;
  }
  __syncthreads();
  for (int i = t; i < CCH * 130; i += 256) {
    int c = i / 130, ml = i % 130, m = m0 + ml;
    float s3 = 0.f;
#pragma unroll
    for (int dn = 0; dn < 3; ++dn) {
      float a = 0.f;
      int base = 2 * ml + dn;
#pragma unroll
      for (int k = 0; k < 16; ++k) a += filt[c * 16 + k] * xs[base + k];
      s3 += a;
    }
    float v = s3 * (1.f / 3.f) + wb[c];
    h[(b * CCH + c) * LPOOL + m] = v * sc[c] + sh[c];
  }
}

// ---- branches: (always|eventually) -> (eventually|always) -> maxpool(4,4) --
// also accumulates BN2 / BN3 channel stats. leaky_relu(relu(x)) == relu(x).
__global__ void k_branches(const float* __restrict__ h,
                           const float* a1_w, const float* a1_b,
                           const float* e1_w, const float* e1_b,
                           const float* f1_w, const float* f1_b,
                           const float* fa_w, const float* fa_b,
                           float* __restrict__ x1c, float* __restrict__ x2c,
                           float* __restrict__ stats) {
  __shared__ float hrow[8 * LPOOL];
  __shared__ float t1[8 * LS1];
  __shared__ float t2[8 * LS2];
  __shared__ float w1s[64], w2s[64];
  __shared__ float off1[8], off2[8], csum[8], csq[8];
  int b = blockIdx.x, c0 = blockIdx.y * 8;
  int t = threadIdx.x;
  for (int j = t; j < 8 * LPOOL; j += 256) {
    int cc = j / LPOOL, l = j % LPOOL;
    hrow[j] = h[(b * CCH + c0 + cc) * LPOOL + l];
  }
  for (int br = 0; br < 2; ++br) {
    const float* W1  = br ? f1_w : a1_w; const float* B1p = br ? f1_b : a1_b;
    const float* W2  = br ? fa_w : e1_w; const float* B2p = br ? fa_b : e1_b;
    __syncthreads();
    if (t < 64) {
      int cc = t >> 3, k = t & 7;
      w1s[t] = W1[(c0 + cc) * 8 + k];
      w2s[t] = W2[(c0 + cc) * 8 + k];
    }
    __syncthreads();
    if (t < 8) {
      float s1 = 0.f, s2 = 0.f;
      for (int k = 0; k < 8; ++k) { s1 += w1s[t * 8 + k]; s2 += w2s[t * 8 + k]; }
      off1[t] = br ? (1.f - B1p[c0 + t]) : (B1p[c0 + t] - s1);  // eventually : always
      off2[t] = br ? (B2p[c0 + t] - s2) : (1.f - B2p[c0 + t]);  // always : eventually
      csum[t] = 0.f; csq[t] = 0.f;
    }
    __syncthreads();
    for (int i = t; i < 8 * LS1; i += 256) {      // stage 1: 520 -> 257
      int cc = i / LS1, n = i % LS1;
      float s = off1[cc];
#pragma unroll
      for (int k = 0; k < 8; ++k) s += hrow[cc * LPOOL + 2 * n + k] * w1s[cc * 8 + k];
      t1[i] = relu_(s);
    }
    __syncthreads();
    for (int i = t; i < 8 * LS2; i += 256) {      // stage 2: 257 -> 125
      int cc = i / LS2, n = i % LS2;
      float s = off2[cc];
#pragma unroll
      for (int k = 0; k < 8; ++k) s += t1[cc * LS1 + 2 * n + k] * w2s[cc * 8 + k];
      t2[i] = relu_(s);
    }
    __syncthreads();
    float* xc = br ? x2c : x1c;
    for (int i = t; i < 8 * LS3; i += 256) {      // maxpool: 125 -> 31
      int cc = i / LS3, m = i % LS3;
      float mx = t2[cc * LS2 + 4 * m];
#pragma unroll
      for (int j = 1; j < 4; ++j) mx = fmaxf(mx, t2[cc * LS2 + 4 * m + j]);
      xc[(b * CCH + c0 + cc) * LS3 + m] = mx;
      atomicAdd(&csum[cc], mx);
      atomicAdd(&csq[cc], mx * mx);
    }
    __syncthreads();
    if (t < 8) {
      int base = br ? 128 : 64;                    // bn2 at 64, bn3 at 128
      atomicAdd(&stats[base + c0 + t], csum[t]);
      atomicAdd(&stats[base + CCH + c0 + t], csq[t]);
    }
  }
}

// ---- BN + final eventually -> z (B,768) f16, concat layout -----------------
__global__ void k_final_pack(const float* __restrict__ x1c, const float* __restrict__ x2c,
                             const float* __restrict__ bn2, const float* __restrict__ bn3,
                             const float* e2_w, const float* e2_b,
                             const float* f2_w, const float* f2_b,
                             _Float16* __restrict__ z) {
  __shared__ float xb[CCH * LS3];
  int b = blockIdx.x, t = threadIdx.x;
  for (int br = 0; br < 2; ++br) {
    const float* xc = br ? x2c : x1c;
    const float* bn = br ? bn3 : bn2;
    const float* W  = br ? f2_w : e2_w;
    const float* Bp = br ? f2_b : e2_b;
    __syncthreads();
    for (int j = t; j < CCH * LS3; j += 256) {
      int c = j / LS3;
      xb[j] = xc[(size_t)b * CCH * LS3 + j] * bn[c] + bn[CCH + c];
    }
    __syncthreads();
    for (int i = t; i < CCH * LS4; i += 256) {
      int c = i / LS4, m = i % LS4;
      float s = 1.f - Bp[c];
#pragma unroll
      for (int k = 0; k < 8; ++k) s += xb[c * LS3 + 2 * m + k] * W[c * 8 + k];
      z[(size_t)b * 768 + br * 384 + i] = (_Float16)relu_(s);
    }
  }
}

// ---- weight transpose + f32 -> f16 (W (K,Nsrc) -> Wt (Ndst,K), zero-pad) ---
__global__ void k_wt(const float* __restrict__ w, _Float16* __restrict__ wt,
                     int K, int Nsrc, int Ndst) {
  int i = blockIdx.x * blockDim.x + threadIdx.x;
  if (i >= Ndst * K) return;
  int n = i / K, k = i % K;
  wt[i] = (_Float16)((n < Nsrc) ? w[(size_t)k * Nsrc + n] : 0.f);
}

// ---- WMMA fragment loads (ISA 7.12.2 layouts) ------------------------------
// A 16x32 f16: lane<16 holds K = {k0..k0+7, k0+16..k0+23}; lane>=16 shifts by 8.
__device__ __forceinline__ v16h load_a_frag(const _Float16* __restrict__ rowPtr,
                                            int k0, int lane) {
  int sel = (lane < 16) ? 0 : 8;
  v8h lo = *(const v8h*)(rowPtr + k0 + sel);
  v8h hi = *(const v8h*)(rowPtr + k0 + sel + 16);
  v16h r;
#pragma unroll
  for (int i = 0; i < 8; ++i) { r[i] = lo[i]; r[i + 8] = hi[i]; }
  return r;
}
// B 32x16 f16 (from Wt row n, length K): lane<16 holds K = {k0..k0+7, k0+16..k0+23}; lane>=16 shifts by 8.
__device__ __forceinline__ v16h load_b_frag(const _Float16* __restrict__ rowPtr,
                                            int k0, int lane) {
  int sel = (lane < 16) ? 0 : 8;
  v8h lo = *(const v8h*)(rowPtr + k0 + sel);
  v8h hi = *(const v8h*)(rowPtr + k0 + sel + 16);
  v16h r;
#pragma unroll
  for (int i = 0; i < 8; ++i) { r[i] = lo[i]; r[i + 8] = hi[i]; }
  return r;
}

// ---- GEMM: out = relu(A (MxK,f16) * Wt^T (NxK,f16) + bias), per-wave 16x16 tile
__global__ void __launch_bounds__(128)
k_gemm_wmma(const _Float16* __restrict__ A, const _Float16* __restrict__ Wt,
            const float* __restrict__ bias,
            _Float16* __restrict__ outH, float* __restrict__ outF,
            int M, int N, int K, int nValid) {
  int lane = threadIdx.x & 31;
  int wid  = blockIdx.x * (blockDim.x >> 5) + (threadIdx.x >> 5);
  int tilesN = N >> 4;
  int tm = wid / tilesN, tn = wid - tm * tilesN;
  if (tm >= (M >> 4)) return;                 // wave-uniform
  int m0 = tm << 4, n0 = tn << 4;
  const _Float16* arow = A  + (size_t)(m0 + (lane & 15)) * K;
  const _Float16* brow = Wt + (size_t)(n0 + (lane & 15)) * K;
  v8f acc = {};
  for (int k0 = 0; k0 < K; k0 += 32) {
    v16h af = load_a_frag(arow, k0, lane);
    v16h bf = load_b_frag(brow, k0, lane);
    acc = __builtin_amdgcn_wmma_f32_16x16x32_f16(false, af, false, bf,
                                                 (short)0, acc, false, false);
  }
  int n  = n0 + (lane & 15);
  int mb = m0 + ((lane >> 4) << 3);           // C/D: VGPR v -> M = v (+8 for upper lanes)
  if (outF) {
    if (n < nValid) {
      float bv = bias[n];
#pragma unroll
      for (int v = 0; v < 8; ++v)
        outF[(size_t)(mb + v) * nValid + n] = relu_(acc[v] + bv);
    }
  } else {
    float bv = bias[n];
#pragma unroll
    for (int v = 0; v < 8; ++v)
      outH[(size_t)(mb + v) * N + n] = (_Float16)relu_(acc[v] + bv);
  }
}

// ---------------------------------------------------------------------------
extern "C" void kernel_launch(void* const* d_in, const int* in_sizes, int n_in,
                              void* d_out, int out_size, void* d_ws, size_t ws_size,
                              hipStream_t stream) {
  (void)in_sizes; (void)n_in; (void)out_size; (void)ws_size;
  const float* x       = (const float*)d_in[0];
  const float* la_a    = (const float*)d_in[1];
  const float* la_b    = (const float*)d_in[2];
  const float* la_bias = (const float*)d_in[3];
  const float* bn1_g   = (const float*)d_in[4];
  const float* bn1_b   = (const float*)d_in[5];
  const float* a1_w    = (const float*)d_in[6];
  const float* a1_b    = (const float*)d_in[7];
  const float* e1_w    = (const float*)d_in[8];
  const float* e1_b    = (const float*)d_in[9];
  const float* bn2_g   = (const float*)d_in[10];
  const float* bn2_b   = (const float*)d_in[11];
  const float* e2_w    = (const float*)d_in[12];
  const float* e2_b    = (const float*)d_in[13];
  const float* f1_w    = (const float*)d_in[14];
  const float* f1_b    = (const float*)d_in[15];
  const float* fa_w    = (const float*)d_in[16];
  const float* fa_b    = (const float*)d_in[17];
  const float* bn3_g   = (const float*)d_in[18];
  const float* bn3_b   = (const float*)d_in[19];
  const float* f2_w    = (const float*)d_in[20];
  const float* f2_b    = (const float*)d_in[21];
  const float* w1 = (const float*)d_in[22]; const float* b1 = (const float*)d_in[23];
  const float* w2 = (const float*)d_in[24]; const float* b2 = (const float*)d_in[25];
  const float* w3 = (const float*)d_in[26]; const float* b3 = (const float*)d_in[27];
  const float* w4 = (const float*)d_in[28]; const float* b4 = (const float*)d_in[29];

  char* ws = (char*)d_ws;
  size_t off = 0;
  auto alloc = [&](size_t bytes) -> char* {
    char* p = ws + off;
    off += (bytes + 255) & ~(size_t)255;
    return p;
  };
  float*     stats = (float*)alloc(192 * 4);  // [0]bn1 sum/sq [64]bn2 [128]bn3
  float*     bn1p  = (float*)alloc(64 * 4);
  float*     bn2p  = (float*)alloc(64 * 4);
  float*     bn3p  = (float*)alloc(64 * 4);
  float*     h     = (float*)alloc((size_t)BATCH * CCH * LPOOL * 4);
  float*     x1c   = (float*)alloc((size_t)BATCH * CCH * LS3 * 4);
  float*     x2c   = (float*)alloc((size_t)BATCH * CCH * LS3 * 4);
  _Float16*  z     = (_Float16*)alloc((size_t)BATCH * 768 * 2);
  _Float16*  z1    = (_Float16*)alloc((size_t)BATCH * 1024 * 2);
  _Float16*  z2    = (_Float16*)alloc((size_t)BATCH * 512 * 2);
  _Float16*  z3    = (_Float16*)alloc((size_t)BATCH * 128 * 2);
  _Float16*  w1t   = (_Float16*)alloc((size_t)1024 * 768 * 2);
  _Float16*  w2t   = (_Float16*)alloc((size_t)512 * 1024 * 2);
  _Float16*  w3t   = (_Float16*)alloc((size_t)128 * 512 * 2);
  _Float16*  w4t   = (_Float16*)alloc((size_t)16 * 128 * 2);

  k_zero<<<1, 192, 0, stream>>>(stats, 192);
  k_wt<<<(1024 * 768 + 255) / 256, 256, 0, stream>>>(w1, w1t, 768, 1024, 1024);
  k_wt<<<(512 * 1024 + 255) / 256, 256, 0, stream>>>(w2, w2t, 1024, 512, 512);
  k_wt<<<(128 * 512 + 255) / 256, 256, 0, stream>>>(w3, w3t, 512, 128, 128);
  k_wt<<<(16 * 128 + 255) / 256, 256, 0, stream>>>(w4, w4t, 128, 10, 16);

  k_conv_stats<<<dim3(BATCH, 9), 256, 0, stream>>>(x, la_a, la_b, la_bias, stats);
  k_bn_finalize<<<1, 32, 0, stream>>>(stats, bn1_g, bn1_b, bn1p,
                                      1.0f / ((float)BATCH * (float)LCONV));
  k_conv_bn_pool<<<dim3(BATCH, 4), 256, 0, stream>>>(x, la_a, la_b, la_bias, bn1p, h);

  k_branches<<<dim3(BATCH, 4), 256, 0, stream>>>(h, a1_w, a1_b, e1_w, e1_b,
                                                 f1_w, f1_b, fa_w, fa_b,
                                                 x1c, x2c, stats);
  k_bn_finalize<<<1, 32, 0, stream>>>(stats + 64,  bn2_g, bn2_b, bn2p,
                                      1.0f / ((float)BATCH * (float)LS3));
  k_bn_finalize<<<1, 32, 0, stream>>>(stats + 128, bn3_g, bn3_b, bn3p,
                                      1.0f / ((float)BATCH * (float)LS3));
  k_final_pack<<<BATCH, 256, 0, stream>>>(x1c, x2c, bn2p, bn3p,
                                          e2_w, e2_b, f2_w, f2_b, z);

  // MLP: each wave computes one 16x16 tile; 4 waves (128 threads) per block.
  k_gemm_wmma<<<(64 * 64) / 4, 128, 0, stream>>>(z,  w1t, b1, z1, nullptr, 1024, 1024, 768, 0);
  k_gemm_wmma<<<(64 * 32) / 4, 128, 0, stream>>>(z1, w2t, b2, z2, nullptr, 1024, 512, 1024, 0);
  k_gemm_wmma<<<(64 * 8)  / 4, 128, 0, stream>>>(z2, w3t, b3, z3, nullptr, 1024, 128, 512, 0);
  k_gemm_wmma<<<(64 * 1)  / 4, 128, 0, stream>>>(z3, w4t, b4, nullptr, (float*)d_out,
                                                 1024, 16, 128, 10);
}
